// RetroModel_3204045603500
// MI455X (gfx1250) — matrix operand order repaired and checked by reading, not compiled, not run on hardware
//
#include <hip/hip_runtime.h>
#include <hip/hip_bf16.h>

// ---------------------------------------------------------------------------
// Model dims (fixed by the reference)
// ---------------------------------------------------------------------------
#define HDIM   64
#define LSEQ   4096
#define BATCH  64
#define HALF   2048
#define VOCABN 128
#define CHUNK  32   // delta-rule chunk size (K dim of the M-update WMMA)

typedef __attribute__((ext_vector_type(16))) __bf16 v16bf;
typedef __attribute__((ext_vector_type(8)))  float  v8f;

// ---------------------------------------------------------------------------
// WMMA fragment loaders (layouts per CDNA5 ISA, wave32)
// ---------------------------------------------------------------------------
// A-matrix 16x32 bf16: lanes 0-15 row M=lane, VGPR v<4 -> K=2v,2v+1 (+8 for
// upper lane half), VGPR v>=4 -> K=16+2(v-4) (+8 for upper half).
__device__ __forceinline__ int a_koff(int v, int hi) {
  return (v < 4) ? (2 * v + (hi ? 8 : 0)) : (2 * (v - 4) + (hi ? 24 : 16));
}
// B-matrix 32x16 bf16: lane n holds column n; same K mapping as the A operand
// (VGPR v<4 -> K=2v,2v+1 (+8 upper half), v>=4 -> K=16+2(v-4) (+8 upper half)).
__device__ __forceinline__ int b_koff(int v, int hi) {
  return (v < 4) ? (2 * v + (hi ? 8 : 0)) : (2 * (v - 4) + (hi ? 24 : 16));
}

// A tile from a row-major bf16 buffer (16 rows, leading dim ld), K offset k0.
__device__ __forceinline__ v16bf load_a(const __bf16* __restrict__ src, int ld,
                                        int lane, int k0) {
  const int m = lane & 15, hi = (lane >> 4) & 1;
  v16bf a;
#pragma unroll
  for (int v = 0; v < 8; ++v) {
    const int kb = k0 + a_koff(v, hi);
    a[2 * v]     = src[m * ld + kb];
    a[2 * v + 1] = src[m * ld + kb + 1];
  }
  return a;
}

// B tile where B[k][n] = W[(n0+n)*ld + k0+k]  (i.e. GEMM with W^T, W row-major)
__device__ __forceinline__ v16bf load_b_nk(const __bf16* __restrict__ W, int ld,
                                           int lane, int n0, int k0) {
  const int n = lane & 15, hi = (lane >> 4) & 1;
  v16bf b;
#pragma unroll
  for (int v = 0; v < 8; ++v) {
    const int kb = k0 + b_koff(v, hi);
    b[2 * v]     = W[(n0 + n) * ld + kb];
    b[2 * v + 1] = W[(n0 + n) * ld + kb + 1];
  }
  return b;
}

// B tile where B[k][n] = S[(k0+k)*ld + n0+n]  (plain KxN row-major operand)
__device__ __forceinline__ v16bf load_b_kn(const __bf16* __restrict__ S, int ld,
                                           int lane, int n0, int k0) {
  const int n = lane & 15, hi = (lane >> 4) & 1;
  v16bf b;
#pragma unroll
  for (int v = 0; v < 8; ++v) {
    const int kb = k0 + b_koff(v, hi);
    b[2 * v]     = S[kb * ld + n0 + n];
    b[2 * v + 1] = S[(kb + 1) * ld + n0 + n];
  }
  return b;
}

#define WMMA_BF16(a, b, c) \
  __builtin_amdgcn_wmma_f32_16x16x32_bf16(false, (a), false, (b), (short)0, (c), false, false)

// ---------------------------------------------------------------------------
// Kernel 0: convert GEMM weights to bf16 once (tiny)
// ---------------------------------------------------------------------------
__global__ void prep_weights_kernel(const float* __restrict__ W1,
                                    const float* __restrict__ W2,
                                    const float* __restrict__ Wk,
                                    const float* __restrict__ Wv,
                                    const float* __restrict__ Wattn,
                                    __bf16* __restrict__ W1bf,
                                    __bf16* __restrict__ W2bf,
                                    __bf16* __restrict__ Wkbf,
                                    __bf16* __restrict__ Wvbf,
                                    __bf16* __restrict__ Wattnbf) {
  const int i = blockIdx.x * blockDim.x + threadIdx.x;
  if (i < 2 * HDIM * HDIM) {            // 8192
    W1bf[i] = (__bf16)W1[i];
    W2bf[i] = (__bf16)W2[i];
  }
  if (i < HDIM * HDIM) {                // 4096
    Wkbf[i]    = (__bf16)Wk[i];
    Wvbf[i]    = (__bf16)Wv[i];
    Wattnbf[i] = (__bf16)Wattn[i];
  }
}

// ---------------------------------------------------------------------------
// Kernel 1: encoder. One wave = 16 tokens. 4 waves / block.
//   embed gather -> FF1(relu) -> FF2 + residual -> LayerNorm ->
//   k = l2norm(h Wk^T), v = h Wv^T. All GEMMs via v_wmma_f32_16x16x32_bf16.
// ---------------------------------------------------------------------------
__global__ __launch_bounds__(128) void encoder_kernel(
    const int* __restrict__ x, const float* __restrict__ embed,
    const float* __restrict__ b1, const float* __restrict__ b2,
    const float* __restrict__ ln_g, const float* __restrict__ ln_b,
    const __bf16* __restrict__ W1bf, const __bf16* __restrict__ W2bf,
    const __bf16* __restrict__ Wkbf, const __bf16* __restrict__ Wvbf,
    __bf16* __restrict__ hiddenBF, __bf16* __restrict__ ksBF,
    __bf16* __restrict__ vsBF) {
  __shared__ float  sF32h[4][16 * HDIM];      // h -> z -> (LN scratch)
  __shared__ float  sF32t[4][16 * HDIM];      // k tile (pre-norm)
  __shared__ __bf16 sBFa[4][16 * HDIM];       // bf16 A operand (h, then hidden)
  __shared__ __bf16 sBFf1[4][16 * 2 * HDIM];  // ff1 activations bf16

  const int lane = threadIdx.x & 31;
  const int wave = threadIdx.x >> 5;
  const long long tok0 = ((long long)blockIdx.x * 4 + wave) * 16;

  float*  hS  = sF32h[wave];
  float*  tS  = sF32t[wave];
  __bf16* aS  = sBFa[wave];
  __bf16* f1S = sBFf1[wave];

  // ---- 1) gather embedding rows -------------------------------------------
  for (int i = lane; i < 16 * HDIM; i += 32) {
    const int t = i >> 6, c = i & 63;
    const int tok = x[tok0 + t];
    const float v = embed[tok * HDIM + c];
    hS[i] = v;
    aS[i] = (__bf16)v;
  }

  const int n  = lane & 15;
  const int hi = lane >> 4;

  // ---- 2) FF1: relu(h @ W1^T + b1) -> f1S bf16 [16][128] -------------------
#pragma unroll
  for (int nt = 0; nt < 8; ++nt) {
    const int n0 = nt * 16;
    v8f acc = {};
#pragma unroll
    for (int kc = 0; kc < 2; ++kc) {
      v16bf a  = load_a(aS, HDIM, lane, kc * 32);
      v16bf bm = load_b_nk(W1bf, HDIM, lane, n0, kc * 32);
      acc = WMMA_BF16(a, bm, acc);
    }
    const float bias = b1[n0 + n];
#pragma unroll
    for (int r = 0; r < 8; ++r) {
      const int m = r + 8 * hi;
      float v = acc[r] + bias;
      v = v > 0.f ? v : 0.f;
      f1S[m * (2 * HDIM) + n0 + n] = (__bf16)v;
    }
  }

  // ---- 3) FF2 + residual: z = h + (f1 @ W2^T + b2)  -> hS ------------------
#pragma unroll
  for (int nt = 0; nt < 4; ++nt) {
    const int n0 = nt * 16;
    v8f acc = {};
#pragma unroll
    for (int kc = 0; kc < 4; ++kc) {
      v16bf a  = load_a(f1S, 2 * HDIM, lane, kc * 32);
      v16bf bm = load_b_nk(W2bf, 2 * HDIM, lane, n0, kc * 32);
      acc = WMMA_BF16(a, bm, acc);
    }
    const float bias = b2[n0 + n];
#pragma unroll
    for (int r = 0; r < 8; ++r) {
      const int m = r + 8 * hi;
      hS[m * HDIM + n0 + n] += acc[r] + bias;
    }
  }

  // ---- 4) LayerNorm; write hidden bf16 to LDS + global ---------------------
  {
    const int row = lane & 15;
    const float* zr = hS + row * HDIM + hi * 32;
    float s = 0.f;
#pragma unroll
    for (int j = 0; j < 32; ++j) s += zr[j];
    s += __shfl_xor(s, 16, 32);
    const float mu = s * (1.f / 64.f);
    float vv = 0.f;
#pragma unroll
    for (int j = 0; j < 32; ++j) { const float d = zr[j] - mu; vv += d * d; }
    vv += __shfl_xor(vv, 16, 32);
    const float rstd = rsqrtf(vv * (1.f / 64.f) + 1e-5f);
#pragma unroll
    for (int j = 0; j < 32; ++j) {
      const int c = hi * 32 + j;
      const float val = (hS[row * HDIM + c] - mu) * rstd * ln_g[c] + ln_b[c];
      const __bf16 bv = (__bf16)val;
      aS[row * HDIM + c] = bv;
      hiddenBF[(tok0 + row) * HDIM + c] = bv;
    }
  }

  // ---- 5) k = l2norm(hidden @ Wk^T) ---------------------------------------
#pragma unroll
  for (int nt = 0; nt < 4; ++nt) {
    const int n0 = nt * 16;
    v8f acc = {};
#pragma unroll
    for (int kc = 0; kc < 2; ++kc) {
      v16bf a  = load_a(aS, HDIM, lane, kc * 32);
      v16bf bm = load_b_nk(Wkbf, HDIM, lane, n0, kc * 32);
      acc = WMMA_BF16(a, bm, acc);
    }
#pragma unroll
    for (int r = 0; r < 8; ++r) tS[(r + 8 * hi) * HDIM + n0 + n] = acc[r];
  }
  {
    const int row = lane & 15;
    float ss = 0.f;
#pragma unroll
    for (int j = 0; j < 32; ++j) {
      const float kv = tS[row * HDIM + hi * 32 + j];
      ss += kv * kv;
    }
    ss += __shfl_xor(ss, 16, 32);
    const float scale = 1.f / fmaxf(sqrtf(ss), 1e-12f);
#pragma unroll
    for (int j = 0; j < 32; ++j) {
      const int c = hi * 32 + j;
      ksBF[(tok0 + row) * HDIM + c] = (__bf16)(tS[row * HDIM + c] * scale);
    }
  }

  // ---- 6) v = hidden @ Wv^T (store straight from accumulators) ------------
#pragma unroll
  for (int nt = 0; nt < 4; ++nt) {
    const int n0 = nt * 16;
    v8f acc = {};
#pragma unroll
    for (int kc = 0; kc < 2; ++kc) {
      v16bf a  = load_a(aS, HDIM, lane, kc * 32);
      v16bf bm = load_b_nk(Wvbf, HDIM, lane, n0, kc * 32);
      acc = WMMA_BF16(a, bm, acc);
    }
#pragma unroll
    for (int r = 0; r < 8; ++r) {
      const int m = r + 8 * hi;
      vsBF[(tok0 + m) * HDIM + n0 + n] = (__bf16)acc[r];
    }
  }
}

// ---------------------------------------------------------------------------
// Kernel 2: chunkwise delta-rule scan (WMMA). One block per batch, 128 thr.
// Exact reformulation over chunks of C=32 steps:
//   A = K_c M^T                    (WMMA, K=64)
//   S = K_c K_c^T                  (WMMA, K=64)
//   u_t = v_t - A[t] - sum_{s<t} S[t,s] u_s    (32-deep serial, 64-col parallel)
//   M  += U^T K_c                  (WMMA, K=32, C-operand accumulate)
// M lives in LDS: f32 master + bf16 mirror for the B operands.
// ---------------------------------------------------------------------------
__global__ __launch_bounds__(128) void delta_scan_kernel(
    const float* __restrict__ Min, const __bf16* __restrict__ ks,
    const __bf16* __restrict__ vs, int steps, float* __restrict__ Mout,
    __bf16* __restrict__ Mbfout) {
  __shared__ float  Mf[HDIM * HDIM];                 // 16KB f32 memory matrix
  __shared__ __bf16 Mb[HDIM * HDIM];                 // 8KB  bf16 mirror
  __shared__ __align__(16) __bf16 Kc[CHUNK * HDIM];  // 4KB  chunk keys
  __shared__ __align__(16) __bf16 Vc[CHUNK * HDIM];  // 4KB  chunk values
  __shared__ float  Af[CHUNK * HDIM];                // 8KB  A = Kc M^T
  __shared__ float  Sf[CHUNK * CHUNK];               // 4KB  S = Kc Kc^T
  __shared__ __bf16 Ut[HDIM * CHUNK];                // 4KB  U^T (row i = col i)

  const int tid  = threadIdx.x;
  const int lane = tid & 31;
  const int wave = tid >> 5;
  const int b    = blockIdx.x;
  const int n    = lane & 15;
  const int hi   = lane >> 4;

  const __bf16* kp = ks + (size_t)b * steps * HDIM;
  const __bf16* vp = vs + (size_t)b * steps * HDIM;

  // init M (zeros for the first scan, seeded for the second)
  for (int i = tid; i < HDIM * HDIM; i += 128) {
    const float m0 = Min ? Min[(size_t)b * HDIM * HDIM + i] : 0.f;
    Mf[i] = m0;
    Mb[i] = (__bf16)m0;
  }
  __syncthreads();

  const int nchunks = steps / CHUNK;
  for (int c = 0; c < nchunks; ++c) {
    const size_t base = (size_t)c * CHUNK * HDIM;

    // ---- stage chunk K/V (4KB each) as 16B vectors -------------------------
    {
      const uint4* k4 = reinterpret_cast<const uint4*>(kp + base);
      const uint4* v4 = reinterpret_cast<const uint4*>(vp + base);
      uint4* K4 = reinterpret_cast<uint4*>(Kc);
      uint4* V4 = reinterpret_cast<uint4*>(Vc);
      for (int i = tid; i < (CHUNK * HDIM) / 8; i += 128) {
        K4[i] = k4[i];
        V4[i] = v4[i];
      }
      if (c + 1 < nchunks) {  // gfx1250 global_prefetch_b8 look-ahead
        __builtin_prefetch(kp + base + CHUNK * HDIM + tid * 32, 0, 0);
        __builtin_prefetch(vp + base + CHUNK * HDIM + tid * 32, 0, 0);
      }
    }
    __syncthreads();

    // ---- A = Kc @ M^T  [32x64]: 8 tiles, 2 per wave ------------------------
#pragma unroll
    for (int p = 0; p < 2; ++p) {
      const int tileid = wave * 2 + p;  // 0..7
      const int rt = tileid >> 2;       // row tile (t)
      const int ct = tileid & 3;        // col tile (i)
      v8f acc = {};
#pragma unroll
      for (int kc2 = 0; kc2 < 2; ++kc2) {
        v16bf a  = load_a(Kc + rt * 16 * HDIM, HDIM, lane, kc2 * 32);
        v16bf bm = load_b_nk(Mb, HDIM, lane, ct * 16, kc2 * 32);
        acc = WMMA_BF16(a, bm, acc);
      }
#pragma unroll
      for (int r = 0; r < 8; ++r)
        Af[(rt * 16 + r + 8 * hi) * HDIM + ct * 16 + n] = acc[r];
    }

    // ---- S = Kc @ Kc^T [32x32]: 4 tiles, 1 per wave ------------------------
    {
      const int rt = wave >> 1, ct = wave & 1;
      v8f acc = {};
#pragma unroll
      for (int kc2 = 0; kc2 < 2; ++kc2) {
        v16bf a  = load_a(Kc + rt * 16 * HDIM, HDIM, lane, kc2 * 32);
        v16bf bm = load_b_nk(Kc + ct * 16 * HDIM, HDIM, lane, 0, kc2 * 32);
        acc = WMMA_BF16(a, bm, acc);
      }
#pragma unroll
      for (int r = 0; r < 8; ++r)
        Sf[(rt * 16 + r + 8 * hi) * CHUNK + ct * 16 + n] = acc[r];
    }
    __syncthreads();

    // ---- forward substitution: u_t = v_t - A[t] - sum_{s<t} S[t,s] u_s -----
    if (tid < HDIM) {  // thread = column j; chain is only CHUNK deep
      float u[CHUNK];
#pragma unroll
      for (int t = 0; t < CHUNK; ++t) {
        float acc = (float)Vc[t * HDIM + tid] - Af[t * HDIM + tid];
#pragma unroll
        for (int s = 0; s < t; ++s) acc -= Sf[t * CHUNK + s] * u[s];
        u[t] = acc;
        Ut[tid * CHUNK + t] = (__bf16)acc;  // store transposed for WMMA A
      }
    }
    __syncthreads();

    // ---- M += U^T @ Kc : 16 tiles (K=32), 4 per wave; C-operand from Mf ----
    {
      const int i0 = wave * 16;
#pragma unroll
      for (int jt = 0; jt < 4; ++jt) {
        const int j0 = jt * 16;
        v8f acc;
#pragma unroll
        for (int r = 0; r < 8; ++r)
          acc[r] = Mf[(i0 + r + 8 * hi) * HDIM + j0 + n];
        v16bf a  = load_a(Ut + i0 * CHUNK, CHUNK, lane, 0);  // A[i][t]=U[t][i]
        v16bf bm = load_b_kn(Kc, HDIM, lane, j0, 0);         // B[t][j]=Kc[t][j]
        acc = WMMA_BF16(a, bm, acc);
#pragma unroll
        for (int r = 0; r < 8; ++r) {
          const int idx = (i0 + r + 8 * hi) * HDIM + j0 + n;
          Mf[idx] = acc[r];
          Mb[idx] = (__bf16)acc[r];
        }
      }
    }
    __syncthreads();
  }

  // ---- write out M ---------------------------------------------------------
  for (int i = tid; i < HDIM * HDIM; i += 128) {
    Mout[(size_t)b * HDIM * HDIM + i] = Mf[i];
    if (Mbfout) Mbfout[(size_t)b * HDIM * HDIM + i] = (__bf16)Mf[i];
  }
}

// ---------------------------------------------------------------------------
// Kernel 3: attention read + second k/v projections. One wave = 16 tokens of
// the first half; all five GEMMs per tile via WMMA; softmax in f32.
// ---------------------------------------------------------------------------
__global__ __launch_bounds__(128) void attn_kernel(
    const __bf16* __restrict__ hiddenBF, const __bf16* __restrict__ Mbf,
    const __bf16* __restrict__ Wattnbf, const __bf16* __restrict__ Wkbf,
    const __bf16* __restrict__ Wvbf, __bf16* __restrict__ ks2BF,
    __bf16* __restrict__ vs2BF) {
  __shared__ __bf16 sBF1[4][16 * HDIM];  // q bf16, later ctx bf16
  __shared__ float  sF[4][16 * HDIM];    // scores f32, later k2 tile
  __shared__ __bf16 sBF2[4][16 * HDIM];  // attn bf16

  const int lane = threadIdx.x & 31;
  const int wave = threadIdx.x >> 5;
  const long long p0 = ((long long)blockIdx.x * 4 + wave) * 16;  // token in half-range
  const int  bb = (int)(p0 >> 11);        // p0 / 2048 (tiles never straddle batches)
  const long long t0 = p0 & (HALF - 1);

  const __bf16* hidBase = hiddenBF + ((size_t)bb * LSEQ + t0) * HDIM;
  const __bf16* Mb      = Mbf + (size_t)bb * HDIM * HDIM;

  __bf16* q1S = sBF1[wave];
  float*  fS  = sF[wave];
  __bf16* atS = sBF2[wave];

  const int n  = lane & 15;
  const int hi = lane >> 4;
  const int row = lane & 15;

  // ---- q = hidden @ Wattn^T ------------------------------------------------
#pragma unroll
  for (int nt = 0; nt < 4; ++nt) {
    const int n0 = nt * 16;
    v8f acc = {};
#pragma unroll
    for (int kc = 0; kc < 2; ++kc) {
      v16bf a  = load_a(hidBase, HDIM, lane, kc * 32);
      v16bf bm = load_b_nk(Wattnbf, HDIM, lane, n0, kc * 32);
      acc = WMMA_BF16(a, bm, acc);
    }
#pragma unroll
    for (int r = 0; r < 8; ++r) q1S[(r + 8 * hi) * HDIM + n0 + n] = (__bf16)acc[r];
  }

  // ---- scores = (q @ M) / 8 -----------------------------------------------
#pragma unroll
  for (int nt = 0; nt < 4; ++nt) {
    const int n0 = nt * 16;
    v8f acc = {};
#pragma unroll
    for (int kc = 0; kc < 2; ++kc) {
      v16bf a  = load_a(q1S, HDIM, lane, kc * 32);
      v16bf bm = load_b_kn(Mb, HDIM, lane, n0, kc * 32);  // B[k][n] = M[k][n]
      acc = WMMA_BF16(a, bm, acc);
    }
#pragma unroll
    for (int r = 0; r < 8; ++r) fS[(r + 8 * hi) * HDIM + n0 + n] = acc[r] * 0.125f;
  }

  // ---- softmax over 64 slots (lane pairs per row) --------------------------
  {
    float mx = -3.4e38f;
#pragma unroll
    for (int j = 0; j < 32; ++j) mx = fmaxf(mx, fS[row * HDIM + hi * 32 + j]);
    mx = fmaxf(mx, __shfl_xor(mx, 16, 32));
    float se = 0.f;
#pragma unroll
    for (int j = 0; j < 32; ++j) {
      const int c = hi * 32 + j;
      const float e = __expf(fS[row * HDIM + c] - mx);
      fS[row * HDIM + c] = e;
      se += e;
    }
    se += __shfl_xor(se, 16, 32);
    const float inv = 1.f / se;
#pragma unroll
    for (int j = 0; j < 32; ++j) {
      const int c = hi * 32 + j;
      atS[row * HDIM + c] = (__bf16)(fS[row * HDIM + c] * inv);
    }
  }

  // ---- ctx = attn @ M^T  (B[k][n] = M[n][k] -> weight-style loader on M) ---
#pragma unroll
  for (int nt = 0; nt < 4; ++nt) {
    const int n0 = nt * 16;
    v8f acc = {};
#pragma unroll
    for (int kc = 0; kc < 2; ++kc) {
      v16bf a  = load_a(atS, HDIM, lane, kc * 32);
      v16bf bm = load_b_nk(Mb, HDIM, lane, n0, kc * 32);
      acc = WMMA_BF16(a, bm, acc);
    }
#pragma unroll
    for (int r = 0; r < 8; ++r)
      q1S[(r + 8 * hi) * HDIM + n0 + n] = (__bf16)acc[r];  // ctx bf16 (q dead)
  }

  // ---- ks2 = l2norm(ctx @ Wk^T) -------------------------------------------
#pragma unroll
  for (int nt = 0; nt < 4; ++nt) {
    const int n0 = nt * 16;
    v8f acc = {};
#pragma unroll
    for (int kc = 0; kc < 2; ++kc) {
      v16bf a  = load_a(q1S, HDIM, lane, kc * 32);
      v16bf bm = load_b_nk(Wkbf, HDIM, lane, n0, kc * 32);
      acc = WMMA_BF16(a, bm, acc);
    }
#pragma unroll
    for (int r = 0; r < 8; ++r) fS[(r + 8 * hi) * HDIM + n0 + n] = acc[r];
  }
  {
    float ss = 0.f;
#pragma unroll
    for (int j = 0; j < 32; ++j) {
      const float kv = fS[row * HDIM + hi * 32 + j];
      ss += kv * kv;
    }
    ss += __shfl_xor(ss, 16, 32);
    const float scale = 1.f / fmaxf(sqrtf(ss), 1e-12f);
#pragma unroll
    for (int j = 0; j < 32; ++j) {
      const int c = hi * 32 + j;
      ks2BF[(p0 + row) * HDIM + c] = (__bf16)(fS[row * HDIM + c] * scale);
    }
  }

  // ---- vs2 = ctx @ Wv^T ----------------------------------------------------
#pragma unroll
  for (int nt = 0; nt < 4; ++nt) {
    const int n0 = nt * 16;
    v8f acc = {};
#pragma unroll
    for (int kc = 0; kc < 2; ++kc) {
      v16bf a  = load_a(q1S, HDIM, lane, kc * 32);
      v16bf bm = load_b_nk(Wvbf, HDIM, lane, n0, kc * 32);
      acc = WMMA_BF16(a, bm, acc);
    }
#pragma unroll
    for (int r = 0; r < 8; ++r) {
      const int m = r + 8 * hi;
      vs2BF[(p0 + m) * HDIM + n0 + n] = (__bf16)acc[r];
    }
  }
}

// ---------------------------------------------------------------------------
// Kernel 4: readout. read = M2 @ (hidden[:,L-1] @ Wq^T); out = read @ Wout^T.
// Tiny (B=64): one block per batch, 64 threads.
// ---------------------------------------------------------------------------
__global__ __launch_bounds__(64) void readout_kernel(
    const __bf16* __restrict__ hiddenBF, const float* __restrict__ M2,
    const float* __restrict__ Wq, const float* __restrict__ Wout,
    const float* __restrict__ bout, float* __restrict__ out) {
  __shared__ float hL[HDIM], qS[HDIM], rS[HDIM];
  const int i = threadIdx.x;
  const int b = blockIdx.x;

  hL[i] = (float)hiddenBF[(((size_t)b * LSEQ) + (LSEQ - 1)) * HDIM + i];
  __syncthreads();

  float q = 0.f;
#pragma unroll
  for (int j = 0; j < HDIM; ++j) q = fmaf(Wq[i * HDIM + j], hL[j], q);
  qS[i] = q;
  __syncthreads();

  float r = 0.f;
#pragma unroll
  for (int j = 0; j < HDIM; ++j)
    r = fmaf(M2[((size_t)b * HDIM + i) * HDIM + j], qS[j], r);
  rS[i] = r;
  __syncthreads();

  for (int v = i; v < VOCABN; v += HDIM) {
    float o = bout[v];
#pragma unroll
    for (int j = 0; j < HDIM; ++j) o = fmaf(Wout[v * HDIM + j], rS[j], o);
    out[(size_t)b * VOCABN + v] = o;
  }
}

// ---------------------------------------------------------------------------
// Host-side launch
// ---------------------------------------------------------------------------
extern "C" void kernel_launch(void* const* d_in, const int* in_sizes, int n_in,
                              void* d_out, int out_size, void* d_ws,
                              size_t ws_size, hipStream_t stream) {
  (void)in_sizes; (void)n_in; (void)out_size; (void)ws_size;

  const int*   x     = (const int*)d_in[0];
  const float* embed = (const float*)d_in[1];
  const float* W1    = (const float*)d_in[2];
  const float* b1    = (const float*)d_in[3];
  const float* W2    = (const float*)d_in[4];
  const float* b2    = (const float*)d_in[5];
  const float* ln_g  = (const float*)d_in[6];
  const float* ln_b  = (const float*)d_in[7];
  const float* Wk    = (const float*)d_in[8];
  const float* Wv    = (const float*)d_in[9];
  const float* Wq    = (const float*)d_in[10];
  const float* Wattn = (const float*)d_in[11];
  const float* Wout  = (const float*)d_in[12];
  const float* bout  = (const float*)d_in[13];
  float* out = (float*)d_out;

  // Workspace carve-up (256B aligned)
  char* ws = (char*)d_ws;
  size_t off = 0;
  auto carve = [&](size_t bytes) -> char* {
    char* p = ws + off;
    off = (off + bytes + 255) & ~(size_t)255;
    return p;
  };
  __bf16* W1bf     = (__bf16*)carve(2 * HDIM * HDIM * sizeof(__bf16));
  __bf16* W2bf     = (__bf16*)carve(2 * HDIM * HDIM * sizeof(__bf16));
  __bf16* Wkbf     = (__bf16*)carve(HDIM * HDIM * sizeof(__bf16));
  __bf16* Wvbf     = (__bf16*)carve(HDIM * HDIM * sizeof(__bf16));
  __bf16* Wattnbf  = (__bf16*)carve(HDIM * HDIM * sizeof(__bf16));
  __bf16* hiddenBF = (__bf16*)carve((size_t)BATCH * LSEQ * HDIM * sizeof(__bf16));
  __bf16* ksBF     = (__bf16*)carve((size_t)BATCH * LSEQ * HDIM * sizeof(__bf16));
  __bf16* vsBF     = (__bf16*)carve((size_t)BATCH * LSEQ * HDIM * sizeof(__bf16));
  __bf16* ks2BF    = (__bf16*)carve((size_t)BATCH * HALF * HDIM * sizeof(__bf16));
  __bf16* vs2BF    = (__bf16*)carve((size_t)BATCH * HALF * HDIM * sizeof(__bf16));
  float*  Mf32     = (float*)carve((size_t)BATCH * HDIM * HDIM * sizeof(float));
  __bf16* Mbf      = (__bf16*)carve((size_t)BATCH * HDIM * HDIM * sizeof(__bf16));
  float*  M2f32    = (float*)carve((size_t)BATCH * HDIM * HDIM * sizeof(float));

  // 0) weights -> bf16
  prep_weights_kernel<<<32, 256, 0, stream>>>(W1, W2, Wk, Wv, Wattn, W1bf, W2bf,
                                              Wkbf, Wvbf, Wattnbf);
  // 1) encoder over all B*L tokens (64 tokens / block)
  encoder_kernel<<<(BATCH * LSEQ) / 64, 128, 0, stream>>>(
      x, embed, b1, b2, ln_g, ln_b, W1bf, W2bf, Wkbf, Wvbf, hiddenBF, ksBF,
      vsBF);
  // 2) first delta scan (M0 = 0), LSEQ steps in 128 chunks of 32
  delta_scan_kernel<<<BATCH, 128, 0, stream>>>(nullptr, ksBF, vsBF, LSEQ, Mf32,
                                               Mbf);
  // 3) attention read + second k/v projections over first half
  attn_kernel<<<(BATCH * HALF) / 64, 128, 0, stream>>>(
      hiddenBF, Mbf, Wattnbf, Wkbf, Wvbf, ks2BF, vs2BF);
  // 4) second delta scan seeded from M, HALF steps in 64 chunks
  delta_scan_kernel<<<BATCH, 128, 0, stream>>>(Mf32, ks2BF, vs2BF, HALF, M2f32,
                                               nullptr);
  // 5) readout
  readout_kernel<<<BATCH, 64, 0, stream>>>(hiddenBF, M2f32, Wq, Wout, bout,
                                           out);
}
